// TABL_DFSMN_81733227643497
// MI455X (gfx1250) — matrix operand order repaired and checked by reading, not compiled, not run on hardware
//
#include <hip/hip_runtime.h>

// ---------------------------------------------------------------------------
// MI455X (gfx1250, wave32) implementation of the DFSMN reference.
//
// Dominant cost: four [8192x2048] @ [2048x2048] GEMMs (~275 GFLOP) -> compute
// bound, mapped onto V_WMMA_F32_16X16X32_BF16 (bf16 in, f32 accumulate).
// B-operands are pre-transposed to bf16 so every WMMA fragment is a
// pair of 16B-per-lane strips (k 8h..+7, 16+8h..+7) matching the CDNA5 16-bit A/B VGPR layouts.
// Tile staging uses GLOBAL_LOAD_ASYNC_TO_LDS_B128 (ASYNCcnt-tracked DMA into
// double-buffered LDS) so data movement overlaps the WMMA stream.
// ---------------------------------------------------------------------------

typedef __bf16 bf16;
typedef __attribute__((ext_vector_type(16))) __bf16 v16bf;
typedef __attribute__((ext_vector_type(4)))  __bf16 v4bf;
typedef __attribute__((ext_vector_type(8)))  float  v8f;
typedef __attribute__((ext_vector_type(4)))  unsigned int uint4v;

union FragBF16 { v16bf v; uint4v u[2]; };

static constexpr int  BN   = 64;    // batch
static constexpr int  DD0  = 256;   // input feature dim
static constexpr int  TT   = 2048;  // time dim
static constexpr int  DD1  = 128;   // hidden dim
static constexpr long long ROWS = (long long)BN * DD1;   // 8192
static constexpr long long SZD  = ROWS * TT;             // 16,777,216 elems

// ---------------------------------------------------------------------------
// f32 -> bf16, 4 elements per thread (n must be a multiple of 4)
// ---------------------------------------------------------------------------
__global__ void f32_to_bf16_vec4(const float* __restrict__ in,
                                 bf16* __restrict__ out, long long n) {
  long long i = ((long long)blockIdx.x * blockDim.x + threadIdx.x) * 4;
  if (i + 3 < n) {
    float4 f = *(const float4*)(in + i);
    v4bf o;
    o[0] = (bf16)f.x; o[1] = (bf16)f.y; o[2] = (bf16)f.z; o[3] = (bf16)f.w;
    *(v4bf*)(out + i) = o;
  }
}

// ---------------------------------------------------------------------------
// Transposing convert: in f32 [R,C] row-major -> out bf16 [C,R] row-major.
// 32x32 LDS tile, 32x8 threads, batched via blockIdx.z.
// ---------------------------------------------------------------------------
__global__ void transpose_to_bf16(const float* __restrict__ in,
                                  bf16* __restrict__ out,
                                  int R, int C,
                                  long long inBatch, long long outBatch) {
  __shared__ float tile[32][33];
  const float* src = in  + (long long)blockIdx.z * inBatch;
  bf16*        dst = out + (long long)blockIdx.z * outBatch;
  int c0 = blockIdx.x * 32, r0 = blockIdx.y * 32;
  int tx = threadIdx.x, ty = threadIdx.y;
  #pragma unroll
  for (int i = ty; i < 32; i += 8)
    tile[i][tx] = src[(long long)(r0 + i) * C + (c0 + tx)];
  __syncthreads();
  #pragma unroll
  for (int i = ty; i < 32; i += 8)
    dst[(long long)(c0 + i) * R + (r0 + tx)] = (bf16)tile[tx][i];
}

// ---------------------------------------------------------------------------
// C[M,N](f32) = A[M,K](bf16, row-major) x Bt[N,K](bf16, row-major = B^T)
// 256 threads = 8 waves (wave32). Block tile 128x128, K-step 32.
// Waves arranged 4 (M) x 2 (N); each wave: 2x4 tiles of 16x16 -> 8 WMMAs/iter.
// K-strips DMA'd into double-buffered LDS with GLOBAL_LOAD_ASYNC_TO_LDS_B128;
// next tile's async loads stay in flight (ASYNCcnt<=4) during the current
// tile's WMMAs. Fragments follow the CDNA5 16-bit A 16x32 / B 32x16 layouts.
// ---------------------------------------------------------------------------
#define KT      32
#define LDSROW  40   // 32 + 8 bf16 pad to spread banks

__global__ __launch_bounds__(256)
void gemm_bf16_wmma(const bf16* __restrict__ A,  long long aBatch,
                    const bf16* __restrict__ Bt, long long bBatch,
                    float* __restrict__ C,       long long cBatch,
                    int Mdim, int Ndim, int Kdim) {
  __shared__ bf16 As[2 * 128 * LDSROW];
  __shared__ bf16 Bs[2 * 128 * LDSROW];
  constexpr unsigned BUFB = 128 * LDSROW * 2;   // bytes per buffer

  A  += (long long)blockIdx.z * aBatch;
  Bt += (long long)blockIdx.z * bBatch;
  C  += (long long)blockIdx.z * cBatch;

  const int nBase = blockIdx.x * 128;
  const int mBase = blockIdx.y * 128;
  const int tid   = threadIdx.x;
  const int lane  = tid & 31;
  const int wid   = tid >> 5;
  const int waveM = wid & 3;          // 0..3  -> 32-row slab
  const int waveN = wid >> 2;         // 0..1  -> 64-col slab
  const int l16   = lane & 15;
  const int khalf = lane >> 4;

  // staging assignment: thread -> (row 0..127, 16-element k-half)
  const int lrow  = tid >> 1;
  const int lhalf = tid & 1;

  v8f c[2][4];
  #pragma unroll
  for (int mi = 0; mi < 2; ++mi)
    #pragma unroll
    for (int ni = 0; ni < 4; ++ni)
      c[mi][ni] = (v8f)0.0f;

  const bf16* gA = A  + (long long)(mBase + lrow) * Kdim + lhalf * 16;
  const bf16* gB = Bt + (long long)(nBase + lrow) * Kdim + lhalf * 16;
  // Low 32 bits of a generic pointer to LDS == allocation-relative LDS address
  const unsigned sA0 = (unsigned)(unsigned long long)&As[lrow * LDSROW + lhalf * 16];
  const unsigned sB0 = (unsigned)(unsigned long long)&Bs[lrow * LDSROW + lhalf * 16];

  // Prologue: DMA tile 0 into buffer 0 (IOFFSET applies to LDS and global).
  asm volatile(
      "global_load_async_to_lds_b128 %0, %2, off\n\t"
      "global_load_async_to_lds_b128 %0, %2, off offset:16\n\t"
      "global_load_async_to_lds_b128 %1, %3, off\n\t"
      "global_load_async_to_lds_b128 %1, %3, off offset:16"
      :: "v"(sA0), "v"(sB0), "v"(gA), "v"(gB) : "memory");
  gA += KT; gB += KT;

  unsigned cur = 0;
  for (int kt = 0; kt < Kdim; kt += KT) {
    if (kt + KT < Kdim) {
      // Issue next tile's DMA into the other buffer, then wait only for the
      // current tile (4 async loads remain in flight during the WMMAs).
      const unsigned dA = sA0 + ((cur ^ 1u) * BUFB);
      const unsigned dB = sB0 + ((cur ^ 1u) * BUFB);
      asm volatile(
          "global_load_async_to_lds_b128 %0, %2, off\n\t"
          "global_load_async_to_lds_b128 %0, %2, off offset:16\n\t"
          "global_load_async_to_lds_b128 %1, %3, off\n\t"
          "global_load_async_to_lds_b128 %1, %3, off offset:16"
          :: "v"(dA), "v"(dB), "v"(gA), "v"(gB) : "memory");
      gA += KT; gB += KT;
      __builtin_prefetch(gA, 0, 3);   // -> global_prefetch_b8 (tile after next)
      __builtin_prefetch(gB, 0, 3);
      asm volatile("s_wait_asynccnt 0x4" ::: "memory");
    } else {
      asm volatile("s_wait_asynccnt 0x0" ::: "memory");
    }
    __syncthreads();                  // all waves' current buffer resident

    const bf16* AsC = &As[cur * 128 * LDSROW];
    const bf16* BsC = &Bs[cur * 128 * LDSROW];

    // A fragments: lane holds row m=l16; VGPR0-3 -> k = khalf*8..+7,
    //              VGPR4-7 -> k = 16 + khalf*8..+7
    FragBF16 fa[2];
    #pragma unroll
    for (int mi = 0; mi < 2; ++mi) {
      const bf16* p = &AsC[(waveM * 32 + mi * 16 + l16) * LDSROW + khalf * 8];
      fa[mi].u[0] = *(const uint4v*)(p);
      fa[mi].u[1] = *(const uint4v*)(p + 16);
    }
    // B fragments: lane holds col n=l16; k = khalf*8..+7, 16 + khalf*8..+7
    FragBF16 fb[4];
    #pragma unroll
    for (int ni = 0; ni < 4; ++ni) {
      const bf16* p = &BsC[(waveN * 64 + ni * 16 + l16) * LDSROW + khalf * 8];
      fb[ni].u[0] = *(const uint4v*)(p);
      fb[ni].u[1] = *(const uint4v*)(p + 16);
    }

    #pragma unroll
    for (int mi = 0; mi < 2; ++mi)
      #pragma unroll
      for (int ni = 0; ni < 4; ++ni)
        c[mi][ni] = __builtin_amdgcn_wmma_f32_16x16x32_bf16(
            false, fa[mi].v, false, fb[ni].v,
            (short)0, c[mi][ni], false, false);

    __syncthreads();                  // all waves done reading buffer `cur`
    cur ^= 1u;
  }

  // C/D layout: lane -> n = l16; VGPR r -> m = r + 8*khalf
  #pragma unroll
  for (int mi = 0; mi < 2; ++mi)
    #pragma unroll
    for (int ni = 0; ni < 4; ++ni) {
      const int n = nBase + waveN * 64 + ni * 16 + l16;
      const int m = mBase + waveM * 32 + mi * 16 + 8 * khalf;
      float* pc = C + (long long)m * Ndim + n;
      #pragma unroll
      for (int r = 0; r < 8; ++r)
        pc[(long long)r * Ndim] = c[mi][ni][r];
    }
}

// ---------------------------------------------------------------------------
// Softmax over batch axis (64 values, kept in registers) + gated memory:
//   A = softmax_b(E);  M = D*A + (first ? 0 : M);  D = D + M   (in place)
// One thread per (o,s) pair: 128*2048 threads; consecutive threads ->
// consecutive s -> coalesced within every batch plane.
// ---------------------------------------------------------------------------
__global__ void softmax_gate(const float* __restrict__ E,
                             float* __restrict__ D,
                             float* __restrict__ M,
                             int first) {
  const int idx = blockIdx.x * blockDim.x + threadIdx.x;   // 0 .. 128*2048-1
  if (idx >= DD1 * TT) return;
  const long long stride = (long long)DD1 * TT;

  float e[BN];
  float mx = -1e30f;
  #pragma unroll
  for (int b = 0; b < BN; ++b) {
    e[b] = E[(long long)b * stride + idx];
    mx = fmaxf(mx, e[b]);
  }
  float s = 0.0f;
  #pragma unroll
  for (int b = 0; b < BN; ++b) {
    e[b] = __expf(e[b] - mx);
    s += e[b];
  }
  const float inv = 1.0f / s;
  #pragma unroll
  for (int b = 0; b < BN; ++b) {
    const long long off = (long long)b * stride + idx;
    const float d = D[off];
    float m = d * (e[b] * inv);
    if (!first) m += M[off];
    M[off] = m;
    D[off] = d + m;
  }
}

// ---------------------------------------------------------------------------
// Y[row, j] = sum_t D[row, t] * W2[t, j] + Bias[row%128, j]   (j = 0..2)
// One block (256 threads) per row; LDS tree reduction.
// ---------------------------------------------------------------------------
__global__ __launch_bounds__(256)
void out_proj(const float* __restrict__ D, const float* __restrict__ W2,
              const float* __restrict__ Bias, float* __restrict__ Y) {
  const int row = blockIdx.x;
  const int o   = row & (DD1 - 1);
  const float* dr = D + (long long)row * TT;
  float a0 = 0.f, a1 = 0.f, a2 = 0.f;
  for (int t = threadIdx.x; t < TT; t += 256) {
    const float d = dr[t];
    a0 += d * W2[t * 3 + 0];
    a1 += d * W2[t * 3 + 1];
    a2 += d * W2[t * 3 + 2];
  }
  __shared__ float red[3][256];
  red[0][threadIdx.x] = a0; red[1][threadIdx.x] = a1; red[2][threadIdx.x] = a2;
  __syncthreads();
  for (int s = 128; s > 0; s >>= 1) {
    if (threadIdx.x < (unsigned)s) {
      red[0][threadIdx.x] += red[0][threadIdx.x + s];
      red[1][threadIdx.x] += red[1][threadIdx.x + s];
      red[2][threadIdx.x] += red[2][threadIdx.x + s];
    }
    __syncthreads();
  }
  if (threadIdx.x == 0) {
    Y[row * 3 + 0] = red[0][0] + Bias[o * 3 + 0];
    Y[row * 3 + 1] = red[1][0] + Bias[o * 3 + 1];
    Y[row * 3 + 2] = red[2][0] + Bias[o * 3 + 2];
  }
}

// ---------------------------------------------------------------------------
extern "C" void kernel_launch(void* const* d_in, const int* in_sizes, int n_in,
                              void* d_out, int out_size, void* d_ws, size_t ws_size,
                              hipStream_t stream) {
  const float* X    = (const float*)d_in[0];   // [64,256,2048]
  const float* W1   = (const float*)d_in[1];   // [128,256]
  const float* DFW[4] = { (const float*)d_in[2], (const float*)d_in[3],
                          (const float*)d_in[4], (const float*)d_in[5] };
  const float* W2   = (const float*)d_in[6];   // [2048,3]
  const float* Bias = (const float*)d_in[7];   // [128,3]
  float* Y = (float*)d_out;                    // [64,128,3]

  // Workspace layout (~232 MiB):
  char* ws = (char*)d_ws;
  float* D    = (float*)(ws);                                   // 64 MiB  [8192,2048] f32
  float* M    = (float*)(ws + (SZD * 4));                       // 64 MiB
  float* E    = (float*)(ws + (SZD * 8));                       // 64 MiB (aliased w/ Xt-bf16)
  bf16*  Abf  = (bf16*) (ws + (SZD * 12));                      // 32 MiB  D in bf16
  bf16*  Wbf  = (bf16*) (ws + (SZD * 12) + (SZD * 2));          //  8 MiB  Wt[s][t] bf16
  bf16*  W1bf = (bf16*) (ws + (SZD * 12) + (SZD * 2) + (long long)TT * TT * 2); // 64 KiB
  bf16*  Xbf  = (bf16*)E;  // Xt[b][t][d] bf16 staging (only live before stage 1)

  // 1) W1 -> bf16 (A-operand, row-major [o,d], contraction contiguous)
  f32_to_bf16_vec4<<<(DD1 * DD0) / 1024, 256, 0, stream>>>(W1, W1bf, DD1 * DD0);

  // 2) X [b][d][t] -> Xt bf16 [b][t][d]  (B-operand transposed)
  transpose_to_bf16<<<dim3(TT / 32, DD0 / 32, BN), dim3(32, 8), 0, stream>>>(
      X, Xbf, DD0, TT, (long long)DD0 * TT, (long long)TT * DD0);

  // 3) X_ = W1 @ X per batch: C[128,2048] = W1[128,256] x Xt[2048,256]^T
  gemm_bf16_wmma<<<dim3(TT / 128, 1, BN), 256, 0, stream>>>(
      W1bf, 0, Xbf, (long long)TT * DD0, D, (long long)DD1 * TT,
      DD1, TT, DD0);

  // 4) Four DFSMN stages
  for (int s = 0; s < 4; ++s) {
    // Wt[s][t] = bf16(W[t][s])
    transpose_to_bf16<<<dim3(TT / 32, TT / 32, 1), dim3(32, 8), 0, stream>>>(
        DFW[s], Wbf, TT, TT, 0, 0);
    // D -> bf16 A-operand
    f32_to_bf16_vec4<<<(int)(SZD / 1024), 256, 0, stream>>>(D, Abf, SZD);
    // E = D @ W : [8192,2048] = [8192,2048] x Wt[2048,2048]^T
    gemm_bf16_wmma<<<dim3(TT / 128, (int)(ROWS / 128), 1), 256, 0, stream>>>(
        Abf, 0, Wbf, 0, E, 0, (int)ROWS, TT, TT);
    // softmax over batch + gated memory update (in place on D, M)
    softmax_gate<<<(DD1 * TT) / 256, 256, 0, stream>>>(E, D, M, s == 0 ? 1 : 0);
  }

  // 5) Y = D @ W2 + Bias
  out_proj<<<(int)ROWS, 256, 0, stream>>>(D, W2, Bias, Y);
}
